// MultiHeadAttention_31061203485333
// MI455X (gfx1250) — matrix operand order repaired and checked by reading, not compiled, not run on hardware
//
#include <hip/hip_runtime.h>
#include <math.h>

// ---------------------------------------------------------------------------
// MI455X (gfx1250) fused multi-head attention forward.
// B=2, S=2048, D_MODEL=512, H=8, d_k=64.
// HBM-bound on the 268MB attn write -> single-pass softmax in the 320KB WGP
// LDS; all matmuls on v_wmma_f32_16x16x32_bf16 (wave32); all fp32->bf16
// conversion hoisted into one bandwidth-bound pre-pass so GEMM inner loops
// are pure vmem + wmma.
// ---------------------------------------------------------------------------

#define BATCH   2
#define SEQ     2048
#define DMODEL  512
#define NHEADS  8
#define DK      64
#define MROWS   (BATCH * SEQ)   // 4096 flattened rows

typedef __attribute__((ext_vector_type(16))) __bf16 v16bf;
typedef __attribute__((ext_vector_type(8)))  float  v8f;

union BF16x16 {
    unsigned int u[8];
    __bf16       h[16];
    v16bf        v;
};

__device__ __forceinline__ v8f wmma_bf16(v16bf a, v16bf b, v8f c) {
    // 8-arg form: (neg_a, A, neg_b, B, c_mod, C, reuse_a, reuse_b)
    return __builtin_amdgcn_wmma_f32_16x16x32_bf16(false, a, false, b,
                                                   (short)0, c, false, false);
}

__device__ __forceinline__ v8f zero8() {
    v8f c = {0.f, 0.f, 0.f, 0.f, 0.f, 0.f, 0.f, 0.f};
    return c;
}

// ---------------------------------------------------------------------------
// Kernel 0: bulk fp32 -> bf16 (native RNE convert, float4 in / uint2 out).
// ---------------------------------------------------------------------------
__global__ __launch_bounds__(256) void cvt_bf16_kernel(
    const float* __restrict__ in, __bf16* __restrict__ out, int n4)
{
    const int i = blockIdx.x * 256 + threadIdx.x;
    if (i < n4) {
        const float4 f = ((const float4*)in)[i];
        union { __bf16 h[4]; uint2 u; } p;
        p.h[0] = (__bf16)f.x;
        p.h[1] = (__bf16)f.y;
        p.h[2] = (__bf16)f.z;
        p.h[3] = (__bf16)f.w;
        ((uint2*)out)[i] = p.u;
    }
}

// ---------------------------------------------------------------------------
// Kernel 1: projection  out = (X @ W^T + bias) * scale  (bf16 in, bf16 out).
// mode 0: head-major store   out[((b*H+h)*S + s)*DK + d]      (Q, K)
// mode 1: transposed store   out[((b*H+h)*DK + d)*S + s]      (V -> Vt)
// ---------------------------------------------------------------------------
__global__ __launch_bounds__(256) void proj_bf16_kernel(
    const __bf16* __restrict__ X, const __bf16* __restrict__ W,
    const float* __restrict__ bias, __bf16* __restrict__ out,
    int mode, float scale)
{
    const int lane = threadIdx.x & 31;
    const int wave = threadIdx.x >> 5;
    const int half = lane >> 4;     // selects K sub-chunk per WMMA layout
    const int l16  = lane & 15;

    const int n0 = blockIdx.x * 128 + wave * 16;
    const int m0 = blockIdx.y * 16;

    const __bf16* arow = X + (size_t)(m0 + l16) * DMODEL;
    const __bf16* brow = W + (size_t)(n0 + l16) * DMODEL;

    v8f c = zero8();
    for (int k0 = 0; k0 < DMODEL; k0 += 32) {
        BF16x16 a, b;
        const __bf16* ap = arow + k0 + half * 8;
        const __bf16* bp = brow + k0 + half * 8;
        __builtin_prefetch(ap + 64, 0, 1);
        __builtin_prefetch(bp + 64, 0, 1);
        *(uint4*)&a.u[0] = *(const uint4*)(ap);
        *(uint4*)&a.u[4] = *(const uint4*)(ap + 16);
        *(uint4*)&b.u[0] = *(const uint4*)(bp);
        *(uint4*)&b.u[4] = *(const uint4*)(bp + 16);
        c = wmma_bf16(a.v, b.v, c);
    }

    const int   n  = n0 + l16;
    const float bv = bias[n];
    const int   h  = n >> 6;      // n / DK
    const int   d  = n & 63;      // n % DK
#pragma unroll
    for (int i = 0; i < 8; ++i) {
        const int m  = m0 + i + half * 8;
        const int bb = m >> 11;   // m / SEQ
        const int s  = m & 2047;  // m % SEQ
        const float val = (c[i] + bv) * scale;
        size_t idx;
        if (mode == 0)
            idx = ((size_t)((bb * NHEADS + h) * SEQ + s)) * DK + d;
        else
            idx = ((size_t)((bb * NHEADS + h) * DK + d)) * SEQ + s;
        out[idx] = (__bf16)val;
    }
}

// ---------------------------------------------------------------------------
// Kernel 2: attention. One workgroup per (b*H+h, 16 q-rows).
// LDS: sc[16][2052] fp32 scores (131 KB, 2052*4 bytes is 16B-aligned/row).
// ---------------------------------------------------------------------------
#define SLD 2052

__global__ __launch_bounds__(256) void attn_kernel(
    const __bf16* __restrict__ Qb, const __bf16* __restrict__ Kb,
    const __bf16* __restrict__ Vt, float* __restrict__ attn_out,
    __bf16* __restrict__ Ob)
{
    extern __shared__ float smem[];
    float* sc     = smem;                 // 16 * SLD
    float* accO   = sc + 16 * SLD;        // 16 * 64
    float* red    = accO + 16 * 64;       // 256
    float* rowmax = red + 256;            // 16
    float* rowinv = rowmax + 16;          // 16

    const int tid  = threadIdx.x;
    const int lane = tid & 31;
    const int wave = tid >> 5;
    const int half = lane >> 4;
    const int l16  = lane & 15;

    const int bh = blockIdx.y;            // b*H + h
    const int q0 = blockIdx.x * 16;

    // ---- Phase 1: scores ----
    const __bf16* Qrow = Qb + ((size_t)bh * SEQ + q0 + l16) * DK;
    BF16x16 a0, a1;                       // Q tile for k0=0 and k0=32 (hoisted)
    {
        const __bf16* ap = Qrow + half * 8;
        *(uint4*)&a0.u[0] = *(const uint4*)(ap);
        *(uint4*)&a0.u[4] = *(const uint4*)(ap + 16);
        *(uint4*)&a1.u[0] = *(const uint4*)(ap + 32);
        *(uint4*)&a1.u[4] = *(const uint4*)(ap + 48);
    }
    for (int jt = wave; jt < SEQ / 16; jt += 8) {
        const int kt = jt * 16;
        const __bf16* Krow = Kb + ((size_t)bh * SEQ + kt + l16) * DK;
        BF16x16 b0, b1;
        {
            const __bf16* bp = Krow + half * 8;
            *(uint4*)&b0.u[0] = *(const uint4*)(bp);
            *(uint4*)&b0.u[4] = *(const uint4*)(bp + 16);
            *(uint4*)&b1.u[0] = *(const uint4*)(bp + 32);
            *(uint4*)&b1.u[4] = *(const uint4*)(bp + 48);
        }
        v8f c = zero8();
        c = wmma_bf16(a0.v, b0.v, c);
        c = wmma_bf16(a1.v, b1.v, c);
#pragma unroll
        for (int i = 0; i < 8; ++i)
            sc[(i + half * 8) * SLD + kt + l16] = c[i];
    }
    __syncthreads();

    // ---- Phase 2: softmax over each of 16 rows (2048 wide) ----
    {
        const int r = tid >> 4, seg = tid & 15;
        const float4* row = (const float4*)(sc + r * SLD + seg * 128);
        float m = -1e30f;
        for (int i = 0; i < 32; ++i) {
            const float4 v = row[i];
            m = fmaxf(m, fmaxf(fmaxf(v.x, v.y), fmaxf(v.z, v.w)));
        }
        red[tid] = m;
    }
    __syncthreads();
    if (tid < 16) {
        float m = red[tid * 16];
        for (int i = 1; i < 16; ++i) m = fmaxf(m, red[tid * 16 + i]);
        rowmax[tid] = m;
    }
    __syncthreads();
    {
        const int r = tid >> 4, seg = tid & 15;
        const float mx = rowmax[r];
        float4* row = (float4*)(sc + r * SLD + seg * 128);
        float ssum = 0.f;
        for (int i = 0; i < 32; ++i) {
            float4 v = row[i];
            v.x = __expf(v.x - mx);
            v.y = __expf(v.y - mx);
            v.z = __expf(v.z - mx);
            v.w = __expf(v.w - mx);
            row[i] = v;
            ssum += v.x + v.y + v.z + v.w;
        }
        red[tid] = ssum;
    }
    __syncthreads();
    if (tid < 16) {
        float s = 0.f;
        for (int i = 0; i < 16; ++i) s += red[tid * 16 + i];
        rowinv[tid] = 1.0f / s;
    }
    __syncthreads();
    // normalize in LDS + single coalesced float4 write of attn (HBM-bound).
    {
        float* attn_base = attn_out + ((size_t)bh * SEQ + q0) * SEQ;
        for (int r = 0; r < 16; ++r) {
            const float inv = rowinv[r];
            float4* row  = (float4*)(sc + r * SLD);
            float4* orow = (float4*)(attn_base + (size_t)r * SEQ);
            for (int i = tid; i < SEQ / 4; i += 256) {
                float4 p = row[i];
                p.x *= inv; p.y *= inv; p.z *= inv; p.w *= inv;
                row[i]  = p;
                orow[i] = p;
            }
        }
        for (int i = tid; i < 16 * 64; i += 256) accO[i] = 0.f;
    }
    __syncthreads();

    // ---- Phase 3: O = P @ V ----
    v8f cacc[4] = {zero8(), zero8(), zero8(), zero8()};
    const int kbase = wave * 256;         // each wave owns a 256-wide k slab
    for (int k0 = kbase; k0 < kbase + 256; k0 += 32) {
        BF16x16 a;
        {
            const float* ap = sc + l16 * SLD + k0 + half * 8;
            const float4 f0 = *(const float4*)(ap);
            const float4 f1 = *(const float4*)(ap + 4);
            const float4 f2 = *(const float4*)(ap + 16);
            const float4 f3 = *(const float4*)(ap + 20);
            a.h[0]  = (__bf16)f0.x; a.h[1]  = (__bf16)f0.y;
            a.h[2]  = (__bf16)f0.z; a.h[3]  = (__bf16)f0.w;
            a.h[4]  = (__bf16)f1.x; a.h[5]  = (__bf16)f1.y;
            a.h[6]  = (__bf16)f1.z; a.h[7]  = (__bf16)f1.w;
            a.h[8]  = (__bf16)f2.x; a.h[9]  = (__bf16)f2.y;
            a.h[10] = (__bf16)f2.z; a.h[11] = (__bf16)f2.w;
            a.h[12] = (__bf16)f3.x; a.h[13] = (__bf16)f3.y;
            a.h[14] = (__bf16)f3.z; a.h[15] = (__bf16)f3.w;
        }
#pragma unroll
        for (int j = 0; j < 4; ++j) {
            BF16x16 b;
            const __bf16* vp =
                Vt + ((size_t)bh * DK + j * 16 + l16) * SEQ + k0 + half * 8;
            *(uint4*)&b.u[0] = *(const uint4*)(vp);
            *(uint4*)&b.u[4] = *(const uint4*)(vp + 16);
            cacc[j] = wmma_bf16(a.v, b.v, cacc[j]);
        }
    }
#pragma unroll
    for (int j = 0; j < 4; ++j)
#pragma unroll
        for (int i = 0; i < 8; ++i)
            atomicAdd(&accO[(i + half * 8) * 64 + j * 16 + l16], cacc[j][i]);
    __syncthreads();

    const int b_ = bh >> 3, h = bh & 7;
    for (int i = tid; i < 16 * 64; i += 256) {
        const int r = i >> 6, d = i & 63;
        Ob[((size_t)(b_ * SEQ + q0 + r)) * DMODEL + h * DK + d] =
            (__bf16)accO[i];
    }
}

// ---------------------------------------------------------------------------
// Kernel 3: y = LN(O @ Wo^T + bo + q_residual). One WG per 16 rows; wave w
// owns cols [w*64, w*64+64). GEMM tile -> LDS row block, fused LN, write y.
// ---------------------------------------------------------------------------
__global__ __launch_bounds__(256) void out_ln_kernel(
    const __bf16* __restrict__ Ob, const __bf16* __restrict__ Wo,
    const float* __restrict__ bo, const float* __restrict__ qin,
    const float* __restrict__ gamma, const float* __restrict__ beta,
    float* __restrict__ y)
{
    __shared__ float xr[16 * DMODEL];     // 32 KB
    __shared__ float red[256];
    __shared__ float mu[16], rstd[16];

    const int tid  = threadIdx.x;
    const int lane = tid & 31;
    const int wave = tid >> 5;
    const int half = lane >> 4;
    const int l16  = lane & 15;

    const int m0 = blockIdx.x * 16;
    const int n0 = wave * 64;

    v8f cacc[4] = {zero8(), zero8(), zero8(), zero8()};
    const __bf16* arow = Ob + (size_t)(m0 + l16) * DMODEL;

    for (int k0 = 0; k0 < DMODEL; k0 += 32) {
        BF16x16 a;
        {
            const __bf16* ap = arow + k0 + half * 8;
            *(uint4*)&a.u[0] = *(const uint4*)(ap);
            *(uint4*)&a.u[4] = *(const uint4*)(ap + 16);
        }
#pragma unroll
        for (int j = 0; j < 4; ++j) {
            BF16x16 b;
            const __bf16* bp =
                Wo + (size_t)(n0 + j * 16 + l16) * DMODEL + k0 + half * 8;
            *(uint4*)&b.u[0] = *(const uint4*)(bp);
            *(uint4*)&b.u[4] = *(const uint4*)(bp + 16);
            cacc[j] = wmma_bf16(a.v, b.v, cacc[j]);
        }
    }

    // epilogue: bias + residual into LDS
#pragma unroll
    for (int j = 0; j < 4; ++j) {
        const int col = n0 + j * 16 + l16;
        const float bv = bo[col];
#pragma unroll
        for (int i = 0; i < 8; ++i) {
            const int r = i + half * 8;
            const int m = m0 + r;
            xr[r * DMODEL + col] =
                cacc[j][i] + bv + qin[(size_t)m * DMODEL + col];
        }
    }
    __syncthreads();

    // LayerNorm: 16 threads per row, each covering 32 cols
    const int r = tid >> 4, seg = tid & 15;
    float s = 0.f, s2 = 0.f;
    {
        const float4* row = (const float4*)(xr + r * DMODEL + seg * 32);
        for (int i = 0; i < 8; ++i) {
            const float4 v = row[i];
            s  += v.x + v.y + v.z + v.w;
            s2 += v.x * v.x + v.y * v.y + v.z * v.z + v.w * v.w;
        }
    }
    red[tid] = s;
    __syncthreads();
    if (tid < 16) {
        float t = 0.f;
        for (int i = 0; i < 16; ++i) t += red[tid * 16 + i];
        mu[tid] = t * (1.0f / DMODEL);
    }
    __syncthreads();
    red[tid] = s2;
    __syncthreads();
    if (tid < 16) {
        float t = 0.f;
        for (int i = 0; i < 16; ++i) t += red[tid * 16 + i];
        const float var = t * (1.0f / DMODEL) - mu[tid] * mu[tid];
        rstd[tid] = rsqrtf(var + 1e-5f);
    }
    __syncthreads();

    for (int rr = 0; rr < 16; ++rr) {
        const float m = mu[rr], rs = rstd[rr];
        const int gm = m0 + rr;
        for (int c = tid; c < DMODEL; c += 256)
            y[(size_t)gm * DMODEL + c] =
                (xr[rr * DMODEL + c] - m) * rs * gamma[c] + beta[c];
    }
}

// ---------------------------------------------------------------------------
extern "C" void kernel_launch(void* const* d_in, const int* in_sizes, int n_in,
                              void* d_out, int out_size, void* d_ws,
                              size_t ws_size, hipStream_t stream)
{
    (void)in_sizes; (void)n_in; (void)out_size; (void)ws_size;

    const float* q     = (const float*)d_in[0];
    const float* k     = (const float*)d_in[1];
    const float* v     = (const float*)d_in[2];
    const float* Wq    = (const float*)d_in[3];
    const float* bq    = (const float*)d_in[4];
    const float* Wk    = (const float*)d_in[5];
    const float* bk    = (const float*)d_in[6];
    const float* Wv    = (const float*)d_in[7];
    const float* bv    = (const float*)d_in[8];
    const float* Wo    = (const float*)d_in[9];
    const float* bo    = (const float*)d_in[10];
    const float* gamma = (const float*)d_in[11];
    const float* beta  = (const float*)d_in[12];

    float* y    = (float*)d_out;
    float* attn = y + (size_t)BATCH * SEQ * DMODEL;   // tuple order: (y, attn)

    // Workspace layout (all __bf16):
    //   Qb | Kb | Vt | Ob     : 4 x 4096*512 = 16 MB
    //   Xq | Xk | Xv          : 3 x 4096*512 = 12 MB
    //   Wqb | Wkb | Wvb | Wob : 4 x 512*512  =  2 MB
    __bf16* Qb  = (__bf16*)d_ws;
    __bf16* Kb  = Qb  + (size_t)MROWS * DMODEL;
    __bf16* Vt  = Kb  + (size_t)MROWS * DMODEL;
    __bf16* Ob  = Vt  + (size_t)MROWS * DMODEL;
    __bf16* Xq  = Ob  + (size_t)MROWS * DMODEL;
    __bf16* Xk  = Xq  + (size_t)MROWS * DMODEL;
    __bf16* Xv  = Xk  + (size_t)MROWS * DMODEL;
    __bf16* Wqb = Xv  + (size_t)MROWS * DMODEL;
    __bf16* Wkb = Wqb + (size_t)DMODEL * DMODEL;
    __bf16* Wvb = Wkb + (size_t)DMODEL * DMODEL;
    __bf16* Wob = Wvb + (size_t)DMODEL * DMODEL;

    const dim3 blk(256);

    // Stage 0: bulk fp32 -> bf16 pre-convert (bandwidth-bound, ~28 MB total).
    const int nX4 = MROWS * DMODEL / 4;    // 524288
    const int nW4 = DMODEL * DMODEL / 4;   // 65536
    cvt_bf16_kernel<<<dim3(nX4 / 256), blk, 0, stream>>>(q,  Xq,  nX4);
    cvt_bf16_kernel<<<dim3(nX4 / 256), blk, 0, stream>>>(k,  Xk,  nX4);
    cvt_bf16_kernel<<<dim3(nX4 / 256), blk, 0, stream>>>(v,  Xv,  nX4);
    cvt_bf16_kernel<<<dim3(nW4 / 256), blk, 0, stream>>>(Wq, Wqb, nW4);
    cvt_bf16_kernel<<<dim3(nW4 / 256), blk, 0, stream>>>(Wk, Wkb, nW4);
    cvt_bf16_kernel<<<dim3(nW4 / 256), blk, 0, stream>>>(Wv, Wvb, nW4);
    cvt_bf16_kernel<<<dim3(nW4 / 256), blk, 0, stream>>>(Wo, Wob, nW4);

    // Stage 1: projections (1/sqrt(d_k)=0.125 folded into Q).
    const dim3 g1(DMODEL / 128, MROWS / 16);
    proj_bf16_kernel<<<g1, blk, 0, stream>>>(Xq, Wqb, bq, Qb, 0, 0.125f);
    proj_bf16_kernel<<<g1, blk, 0, stream>>>(Xk, Wkb, bk, Kb, 0, 1.0f);
    proj_bf16_kernel<<<g1, blk, 0, stream>>>(Xv, Wvb, bv, Vt, 1, 1.0f);

    // Stage 2: attention. 136.6 KB dynamic LDS per WG (2 WGs / 320KB WGP).
    const size_t smem2 =
        (size_t)(16 * SLD + 16 * 64 + 256 + 32) * sizeof(float);
    attn_kernel<<<dim3(SEQ / 16, BATCH * NHEADS), blk, smem2, stream>>>(
        Qb, Kb, Vt, attn, Ob);

    // Stage 3: output projection + residual + LayerNorm.
    out_ln_kernel<<<dim3(MROWS / 16), blk, 0, stream>>>(
        Ob, Wob, bo, q, gamma, beta, y);
}
